// AdvancedTransformer_87170656240271
// MI455X (gfx1250) — matrix operand order repaired and checked by reading, not compiled, not run on hardware
//
#include <hip/hip_runtime.h>
#include <hip/hip_bf16.h>

// ---------------------------------------------------------------------------
// AdvancedTransformer forward for MI455X (gfx1250, wave32, WMMA).
// f16 data path: weights pre-converted (and pre-transposed to [N,K]) once per
// launch; activations kept in f16 wherever they feed a GEMM A-operand.
// GEMM: 8 waves/block, each wave owns a 32x64 register tile = 8 chained
// v_wmma_f32_16x16x32_f16 per 32-wide k-step, pure b128 f16 loads.
// Banded attention: flash-style, scores in LDS (f32), probs in LDS (f16),
// V consumed in transposed [HID, S] layout for contiguous fragments.
// ---------------------------------------------------------------------------

typedef __attribute__((ext_vector_type(16))) _Float16 v16h;
typedef __attribute__((ext_vector_type(8)))  _Float16 v8h;
typedef __attribute__((ext_vector_type(8)))  float    v8f;

#define S_LEN    2048
#define HIDC     512
#define NHEAD    8
#define HDIM     64
#define HALF_WIN 256
#define LDSW     544            // max band width (528) rounded up to 32 + pad
#define VOCAB_N  32000
#define SLOTS_N  256
#define MDIM_N   512

__device__ __forceinline__ v16h make_frag(const _Float16* p0, const _Float16* p1)
{
  const v8h a = *(const v8h*)p0;
  const v8h b = *(const v8h*)p1;
  v16h r;
#pragma unroll
  for (int i = 0; i < 8; ++i) { r[i] = a[i]; r[i + 8] = b[i]; }
  return r;
}

#define WMMA(Af, Bf, Cc) __builtin_amdgcn_wmma_f32_16x16x32_f16( \
    false, (Af), false, (Bf), (short)0, (Cc), false, false)

// ---------------------------------------------------------------------------
// f16 WMMA GEMM: C = act(A[M,K] @ WT^T + bias) (+resid)
//   A  : f16 [M,K] row-major
//   WT : f16 [N,K] row-major (pre-transposed weight)
// flags: 1=ReLU, 2=store f32 C, 4=store f16 C16 [M,N], 8=store f16 C16 [N,M]
// block = 256 (8 waves, 2x4), wave tile = 32x64, block tile = 64x256
// grid = (N/256, M/64)
// ---------------------------------------------------------------------------
__global__ __launch_bounds__(256) void wmma_gemm_h_kernel(
    const _Float16* __restrict__ A, const _Float16* __restrict__ WT,
    const float* __restrict__ bias, const float* __restrict__ resid,
    float* __restrict__ C, _Float16* __restrict__ C16,
    int M, int K, int N, int flags)
{
  const int lane = threadIdx.x & 31;
  const int wave = threadIdx.x >> 5;
  const int wm = wave >> 2, wn = wave & 3;
  const int lo = lane & 15, hi = lane >> 4;
  const int m0 = blockIdx.y * 64 + wm * 32;
  const int n0 = blockIdx.x * 256 + wn * 64;

  const _Float16* a0p = A + (size_t)(m0 + lo) * K;
  const _Float16* a1p = A + (size_t)(m0 + 16 + lo) * K;
  const _Float16* b0p = WT + (size_t)(n0 +  0 + lo) * K;
  const _Float16* b1p = WT + (size_t)(n0 + 16 + lo) * K;
  const _Float16* b2p = WT + (size_t)(n0 + 32 + lo) * K;
  const _Float16* b3p = WT + (size_t)(n0 + 48 + lo) * K;

  const int ka = 16 * hi;   // A fragment: lanes 16-31 hold K=16..31
  const int kb = 16 * hi;   // B fragment: lanes 16-31 hold K=16..31

  v8f acc00 = {}, acc01 = {}, acc02 = {}, acc03 = {};
  v8f acc10 = {}, acc11 = {}, acc12 = {}, acc13 = {};

  for (int k0 = 0; k0 < K; k0 += 32) {
    __builtin_prefetch(a0p + k0 + 128, 0, 1);   // global_prefetch_b8
    __builtin_prefetch(b0p + k0 + 128, 0, 1);
    const v16h af0 = make_frag(a0p + k0 + ka, a0p + k0 + ka + 8);
    const v16h af1 = make_frag(a1p + k0 + ka, a1p + k0 + ka + 8);
    const v16h bf0 = make_frag(b0p + k0 + kb, b0p + k0 + kb + 8);
    const v16h bf1 = make_frag(b1p + k0 + kb, b1p + k0 + kb + 8);
    const v16h bf2 = make_frag(b2p + k0 + kb, b2p + k0 + kb + 8);
    const v16h bf3 = make_frag(b3p + k0 + kb, b3p + k0 + kb + 8);
    acc00 = WMMA(af0, bf0, acc00);  acc10 = WMMA(af1, bf0, acc10);
    acc01 = WMMA(af0, bf1, acc01);  acc11 = WMMA(af1, bf1, acc11);
    acc02 = WMMA(af0, bf2, acc02);  acc12 = WMMA(af1, bf2, acc12);
    acc03 = WMMA(af0, bf3, acc03);  acc13 = WMMA(af1, bf3, acc13);
  }

  v8f* accs[2][4] = {{&acc00, &acc01, &acc02, &acc03},
                     {&acc10, &acc11, &acc12, &acc13}};
#pragma unroll
  for (int j = 0; j < 2; ++j) {
#pragma unroll
    for (int t = 0; t < 4; ++t) {
      const int ncol = n0 + t * 16 + lo;
      const float bv = bias ? bias[ncol] : 0.0f;
      const v8f a = *accs[j][t];
#pragma unroll
      for (int r = 0; r < 8; ++r) {
        const int m = m0 + j * 16 + r + 8 * hi;   // C layout: halves +8 rows
        float v = a[r] + bv;
        if (flags & 1) v = fmaxf(v, 0.0f);
        if (resid) v += resid[(size_t)m * N + ncol];
        if (flags & 2) C[(size_t)m * N + ncol] = v;
        if (flags & 4) C16[(size_t)m * N + ncol] = (_Float16)v;
        if (flags & 8) C16[(size_t)ncol * M + m] = (_Float16)v;  // transposed
      }
    }
  }
}

// ---------------------------------------------------------------------------
// Banded flash attention: one wave per (16-row q-tile, head).
// Q,K: f16 [S,512]; Vt: f16 [512,S] (transposed); O16: f16 [S,512].
// grid = (S/16, NHEAD), block = 32.
// ---------------------------------------------------------------------------
__global__ __launch_bounds__(32) void attn_band_kernel(
    const _Float16* __restrict__ Q, const _Float16* __restrict__ Km,
    const _Float16* __restrict__ Vt, _Float16* __restrict__ O16)
{
  __shared__ float    sc[16 * LDSW];
  __shared__ _Float16 sp[16 * LDSW];
  __shared__ float    sinv[16];

  const int qt   = blockIdx.x;
  const int h    = blockIdx.y;
  const int lane = threadIdx.x;
  const int lo   = lane & 15;
  const int hi   = lane >> 4;
  const int q0   = qt * 16;
  const int ks   = (q0 - HALF_WIN > 0) ? (q0 - HALF_WIN) : 0;
  const int ke   = (q0 + 16 + HALF_WIN < S_LEN) ? (q0 + 16 + HALF_WIN) : S_LEN;
  const int width   = ke - ks;            // multiple of 16
  const int ntiles  = width >> 4;
  const int width32 = (width + 31) & ~31;
  const int cbase   = h * HDIM;

  // Q fragments (two 32-wide d-chunks)
  const _Float16* qrow = Q + (size_t)(q0 + lo) * HIDC + cbase;
  const v16h aq0 = make_frag(qrow + 16 * hi,      qrow + 16 * hi + 8);
  const v16h aq1 = make_frag(qrow + 32 + 16 * hi, qrow + 32 + 16 * hi + 8);

  // scores -> LDS (scale 1/sqrt(64) folded in, band-masked)
  for (int t = 0; t < ntiles; ++t) {
    const int kb0 = ks + t * 16;
    const _Float16* krow = Km + (size_t)(kb0 + lo) * HIDC + cbase;
    const v16h bk0 = make_frag(krow + 16 * hi,      krow + 16 * hi + 8);
    const v16h bk1 = make_frag(krow + 32 + 16 * hi, krow + 32 + 16 * hi + 8);
    v8f acc = {};
    acc = WMMA(aq0, bk0, acc);
    acc = WMMA(aq1, bk1, acc);
#pragma unroll
    for (int r = 0; r < 8; ++r) {
      const int lrow = r + 8 * hi;
      const int qi = q0 + lrow;
      const int kj = kb0 + lo;
      int d = qi - kj; if (d < 0) d = -d;
      sc[lrow * LDSW + t * 16 + lo] = (d <= HALF_WIN) ? 0.125f * acc[r] : -1e30f;
    }
  }
  if (width32 != width) {           // pad to multiple of 32 keys
    for (int r = hi; r < 16; r += 2)
      sc[r * LDSW + width + lo] = -1e30f;
  }
  __syncthreads();

  // softmax: row = lo, lanes hi=0/1 split columns; probs -> f16 LDS
  {
    const int W2 = width32 >> 1;
    const int row = lo;
    float m = -1e30f;
    for (int c = hi * W2; c < (hi + 1) * W2; ++c)
      m = fmaxf(m, sc[row * LDSW + c]);
    m = fmaxf(m, __shfl_xor(m, 16, 32));
    float s = 0.0f;
    for (int c = hi * W2; c < (hi + 1) * W2; ++c) {
      const float e = __expf(sc[row * LDSW + c] - m);
      sp[row * LDSW + c] = (_Float16)e;      // un-normalized probs
      s += e;
    }
    s += __shfl_xor(s, 16, 32);
    if (hi == 0) sinv[row] = 1.0f / s;
  }
  __syncthreads();

  // ctx = P @ V : 32 keys per step, 4 output d-chunks of 16
  v8f c0 = {}, c1 = {}, c2 = {}, c3 = {};
  for (int t2 = 0; t2 < (width32 >> 5); ++t2) {
    const _Float16* prow = sp + lo * LDSW + t2 * 32;
    const v16h ap = make_frag(prow + 16 * hi, prow + 16 * hi + 8);  // ds_load_b128

    int kb2a = ks + t2 * 32 + 16 * hi;
    int kb2b = kb2a + 8;
    if (kb2a > S_LEN - 8) kb2a = S_LEN - 8;   // padded probs are 0 anyway
    if (kb2b > S_LEN - 8) kb2b = S_LEN - 8;
    const _Float16* v0 = Vt + (size_t)(cbase +  0 + lo) * S_LEN;
    const _Float16* v1 = Vt + (size_t)(cbase + 16 + lo) * S_LEN;
    const _Float16* v2 = Vt + (size_t)(cbase + 32 + lo) * S_LEN;
    const _Float16* v3 = Vt + (size_t)(cbase + 48 + lo) * S_LEN;
    c0 = WMMA(ap, make_frag(v0 + kb2a, v0 + kb2b), c0);
    c1 = WMMA(ap, make_frag(v1 + kb2a, v1 + kb2b), c1);
    c2 = WMMA(ap, make_frag(v2 + kb2a, v2 + kb2b), c2);
    c3 = WMMA(ap, make_frag(v3 + kb2a, v3 + kb2b), c3);
  }

#pragma unroll
  for (int r = 0; r < 8; ++r) {
    const int lrow = r + 8 * hi;
    const float inv = sinv[lrow];
    _Float16* orow = O16 + (size_t)(q0 + lrow) * HIDC + cbase;
    orow[ 0 + lo] = (_Float16)(c0[r] * inv);
    orow[16 + lo] = (_Float16)(c1[r] * inv);
    orow[32 + lo] = (_Float16)(c2[r] * inv);
    orow[48 + lo] = (_Float16)(c3[r] * inv);
  }
}

// ---------------------------------------------------------------------------
// Weight conversion kernels (run once at launch head, stream-ordered)
// ---------------------------------------------------------------------------
__global__ __launch_bounds__(256) void cvt_f16_kernel(
    const float* __restrict__ src, _Float16* __restrict__ dst, int n)
{
  for (int i = blockIdx.x * blockDim.x + threadIdx.x; i < n; i += gridDim.x * blockDim.x)
    dst[i] = (_Float16)src[i];
}

// src [K,N] row-major -> dst [N,K] row-major (f16)
__global__ __launch_bounds__(256) void cvt_f16_t_kernel(
    const float* __restrict__ src, _Float16* __restrict__ dst, int K, int N)
{
  const int total = K * N;
  for (int i = blockIdx.x * blockDim.x + threadIdx.x; i < total; i += gridDim.x * blockDim.x) {
    const int k = i / N, n = i - k * N;
    dst[(size_t)n * K + k] = (_Float16)src[i];
  }
}

// ---------------------------------------------------------------------------
// Small wave32 helper kernels
// ---------------------------------------------------------------------------
__global__ __launch_bounds__(256) void embed_kernel(
    const int* __restrict__ ids, const float* __restrict__ tok,
    const float* __restrict__ pos, float* __restrict__ X)
{
  const int row = blockIdx.x;
  const int id = ids[row];
  for (int c = threadIdx.x; c < HIDC; c += blockDim.x)
    X[(size_t)row * HIDC + c] =
        tok[(size_t)id * HIDC + c] + pos[(size_t)row * HIDC + c];
}

// LayerNorm: f32 in -> f16 out (feeds GEMM A operands only)
__global__ __launch_bounds__(32) void layernorm_kernel(
    const float* __restrict__ X, const float* __restrict__ g,
    const float* __restrict__ b, _Float16* __restrict__ Y)
{
  const int row = blockIdx.x, lane = threadIdx.x;
  float s = 0.0f, s2 = 0.0f;
  for (int c = lane; c < HIDC; c += 32) {
    const float v = X[(size_t)row * HIDC + c];
    s += v; s2 += v * v;
  }
  for (int o = 16; o; o >>= 1) { s += __shfl_xor(s, o, 32); s2 += __shfl_xor(s2, o, 32); }
  const float mean = s * (1.0f / HIDC);
  const float var  = s2 * (1.0f / HIDC) - mean * mean;
  const float rstd = rsqrtf(var + 1e-5f);
  for (int c = lane; c < HIDC; c += 32) {
    const float v = (X[(size_t)row * HIDC + c] - mean) * rstd;
    Y[(size_t)row * HIDC + c] = (_Float16)(v * g[c] + b[c]);
  }
}

// row softmax (memory scores): f32 in-place + f16 copy for next GEMM
__global__ __launch_bounds__(32) void row_softmax_kernel(
    float* __restrict__ X, _Float16* __restrict__ X16, int N)
{
  const int row = blockIdx.x, lane = threadIdx.x;
  float m = -1e30f;
  for (int c = lane; c < N; c += 32) m = fmaxf(m, X[(size_t)row * N + c]);
  for (int o = 16; o; o >>= 1) m = fmaxf(m, __shfl_xor(m, o, 32));
  float s = 0.0f;
  for (int c = lane; c < N; c += 32) {
    const float e = __expf(X[(size_t)row * N + c] - m);
    X[(size_t)row * N + c] = e; s += e;
  }
  for (int o = 16; o; o >>= 1) s += __shfl_xor(s, o, 32);
  const float inv = 1.0f / s;
  for (int c = lane; c < N; c += 32) {
    const float p = X[(size_t)row * N + c] * inv;
    X[(size_t)row * N + c] = p;
    X16[(size_t)row * N + c] = (_Float16)p;
  }
}

// concat [x | read] -> f16 (feeds mem-out GEMM)
__global__ __launch_bounds__(256) void concat_kernel(
    const float* __restrict__ A, const float* __restrict__ Bm,
    _Float16* __restrict__ Cc)
{
  const int row = blockIdx.x;
  for (int c = threadIdx.x; c < 2 * HIDC; c += blockDim.x)
    Cc[(size_t)row * (2 * HIDC) + c] = (_Float16)(
        (c < HIDC) ? A[(size_t)row * HIDC + c] : Bm[(size_t)row * HIDC + (c - HIDC)]);
}

// ACT halting: pr = sigmoid(G1[row]·h2w + h2b); update hp / n_upd; emit p_adj
__global__ __launch_bounds__(32) void act_halt_kernel(
    const float* __restrict__ G1, const float* __restrict__ h2w,
    const float* __restrict__ h2b, float* __restrict__ hp,
    float* __restrict__ nupd, float* __restrict__ padj)
{
  const int row = blockIdx.x, lane = threadIdx.x;
  float s = 0.0f;
  for (int i = lane; i < 256; i += 32) s += G1[(size_t)row * 256 + i] * h2w[i];
  for (int o = 16; o; o >>= 1) s += __shfl_xor(s, o, 32);
  if (lane == 0) {
    const float pr = 1.0f / (1.0f + __expf(-(s + h2b[0])));
    const float h = hp[row];
    const float still = (h < 0.99f) ? 1.0f : 0.0f;
    const float nh = (((h + pr) > 0.99f) ? 1.0f : 0.0f) * still;
    const float pa = pr * still + nh * (0.99f - h);
    hp[row] = h + pa;
    nupd[row] += still;
    padj[row] = pa;
  }
}

__global__ __launch_bounds__(256) void act_accum_kernel(
    const float* __restrict__ proc, const float* __restrict__ padj,
    float* __restrict__ out)
{
  const int row = blockIdx.x;
  const float pa = padj[row];
  for (int c = threadIdx.x; c < HIDC; c += blockDim.x)
    out[(size_t)row * HIDC + c] += proc[(size_t)row * HIDC + c] * pa;
}

__global__ __launch_bounds__(256) void ponder_kernel(
    const float* __restrict__ NU, float* __restrict__ out)
{
  __shared__ float red[256];
  float s = 0.0f;
  for (int i = threadIdx.x; i < S_LEN; i += 256) s += NU[i];
  red[threadIdx.x] = s;
  __syncthreads();
  for (int o = 128; o; o >>= 1) {
    if (threadIdx.x < o) red[threadIdx.x] += red[threadIdx.x + o];
    __syncthreads();
  }
  if (threadIdx.x == 0) out[0] = red[0] * (1.0f / S_LEN);
}

// ---------------------------------------------------------------------------
// Launch. Input order = setup_inputs() dict insertion order, recursive
// (lin dicts are {w, b}):
// 0 input_ids | 1 tok_emb | 2 pos_emb
// 3 + 16*L : q.w q.b k.w k.b v.w v.b o.w o.b ff1.w ff1.b ff2.w ff2.b
//            ln1_g ln1_b ln2_g ln2_b
// 99 memory | 100 rq.w 101 rq.b | 102 mout.w 103 mout.b
// 104..111 act: h1.w h1.b h2.w h2.b p1.w p1.b p2.w p2.b
// 112 out.w 113 out.b
// ---------------------------------------------------------------------------
extern "C" void kernel_launch(void* const* d_in, const int* in_sizes, int n_in,
                              void* d_out, int out_size, void* d_ws, size_t ws_size,
                              hipStream_t stream)
{
  (void)in_sizes; (void)n_in; (void)out_size; (void)ws_size;
  const int* ids = (const int*)d_in[0];
  auto F = [&](int i) { return (const float*)d_in[i]; };
  const float* tok = F(1);
  const float* pos = F(2);
  const float* mem   = F(99);
  const float* rqw   = F(100); const float* rqb   = F(101);
  const float* moutw = F(102); const float* moutb = F(103);
  const float* h1w = F(104); const float* h1b = F(105);
  const float* h2w = F(106); const float* h2b = F(107);
  const float* p1w = F(108); const float* p1b = F(109);
  const float* p2w = F(110); const float* p2b = F(111);
  const float* outw = F(112); const float* outb = F(113);

  // ---- workspace: f32 region then f16 region ----
  float* wsF = (float*)d_ws;
  size_t of = 0;
  float* X    = wsF + of; of += (size_t)S_LEN * HIDC;      // residual stream
  float* RD   = wsF + of; of += (size_t)S_LEN * MDIM_N;    // memory read
  float* MS   = wsF + of; of += (size_t)S_LEN * SLOTS_N;   // memory scores
  float* G1   = wsF + of; of += (size_t)S_LEN * 256;       // ACT h1 hidden
  float* CbF  = wsF + of; of += (size_t)S_LEN * HIDC;      // ACT output acc
  float* PROC = wsF + of; of += (size_t)S_LEN * HIDC;      // ACT processed f32
  float* HP   = wsF + of; of += S_LEN;
  float* NU   = wsF + of; of += S_LEN;
  float* PA   = wsF + of; of += S_LEN;

  _Float16* wsH = (_Float16*)(wsF + of);
  size_t oh = 0;
  auto allocH = [&](size_t n) { _Float16* p = wsH + oh; oh += n; return p; };
  // activations (f16)
  _Float16* X2h  = allocH((size_t)S_LEN * HIDC);       // LN out
  _Float16* Qh   = allocH((size_t)S_LEN * HIDC);
  _Float16* Kh   = allocH((size_t)S_LEN * HIDC);
  _Float16* Vth  = allocH((size_t)HIDC * S_LEN);       // V transposed [HID,S]
  _Float16* Ch   = allocH((size_t)S_LEN * HIDC);       // attention ctx
  _Float16* FFh  = allocH((size_t)S_LEN * 4 * HIDC);   // FFN / p1 hidden
  _Float16* Xh   = allocH((size_t)S_LEN * HIDC);       // f16 shadow of X
  _Float16* MSh  = allocH((size_t)S_LEN * SLOTS_N);    // memory probs f16
  _Float16* CATh = allocH((size_t)S_LEN * 2 * HIDC);   // [x|read] f16
  _Float16* Sh0  = allocH((size_t)S_LEN * HIDC);       // ACT state ping
  _Float16* Sh1  = allocH((size_t)S_LEN * HIDC);       // ACT state pong
  _Float16* Cbh  = allocH((size_t)S_LEN * HIDC);       // ACT output f16
  // pre-converted weights (f16, [N,K] transposed unless noted)
  _Float16* qwT[6]; _Float16* kwT[6]; _Float16* vwT[6]; _Float16* owT[6];
  _Float16* f1T[6]; _Float16* f2T[6];
  for (int L = 0; L < 6; ++L) {
    qwT[L] = allocH((size_t)HIDC * HIDC);
    kwT[L] = allocH((size_t)HIDC * HIDC);
    vwT[L] = allocH((size_t)HIDC * HIDC);
    owT[L] = allocH((size_t)HIDC * HIDC);
    f1T[L] = allocH((size_t)4 * HIDC * HIDC);
    f2T[L] = allocH((size_t)4 * HIDC * HIDC);
  }
  _Float16* memNT = allocH((size_t)SLOTS_N * MDIM_N);  // memory as-is  [256,512]
  _Float16* memT  = allocH((size_t)MDIM_N * SLOTS_N);  // memory^T      [512,256]
  _Float16* rqT   = allocH((size_t)MDIM_N * HIDC);
  _Float16* moutT = allocH((size_t)HIDC * 2 * HIDC);
  _Float16* h1T   = allocH((size_t)256 * HIDC);
  _Float16* p1T   = allocH((size_t)2 * HIDC * HIDC);
  _Float16* p2T   = allocH((size_t)HIDC * 2 * HIDC);
  _Float16* outT  = allocH((size_t)VOCAB_N * HIDC);

  auto cvtT = [&](const float* src, _Float16* dst, int K, int N) {
    const int n = K * N;
    cvt_f16_t_kernel<<<(n + 255) / 256, 256, 0, stream>>>(src, dst, K, N);
  };
  auto cvt = [&](const float* src, _Float16* dst, int n) {
    cvt_f16_kernel<<<(n + 255) / 256, 256, 0, stream>>>(src, dst, n);
  };

  // ---- weight conversion pass ----
  for (int L = 0; L < 6; ++L) {
    const int b = 3 + L * 16;
    cvtT(F(b + 0), qwT[L], HIDC, HIDC);
    cvtT(F(b + 2), kwT[L], HIDC, HIDC);
    cvtT(F(b + 4), vwT[L], HIDC, HIDC);
    cvtT(F(b + 6), owT[L], HIDC, HIDC);
    cvtT(F(b + 8), f1T[L], HIDC, 4 * HIDC);
    cvtT(F(b + 10), f2T[L], 4 * HIDC, HIDC);
  }
  cvt (mem,   memNT, SLOTS_N * MDIM_N);
  cvtT(mem,   memT,  SLOTS_N, MDIM_N);   // src [K=256 slots, N=512] -> [512,256]
  cvtT(rqw,   rqT,   HIDC, MDIM_N);
  cvtT(moutw, moutT, 2 * HIDC, HIDC);
  cvtT(h1w,   h1T,   HIDC, 256);
  cvtT(p1w,   p1T,   HIDC, 2 * HIDC);
  cvtT(p2w,   p2T,   2 * HIDC, HIDC);
  cvtT(outw,  outT,  HIDC, VOCAB_N);

  auto gemm = [&](const _Float16* A, const _Float16* WT, const float* B,
                  const float* R, float* Co, _Float16* Co16,
                  int M, int Kd, int Nd, int flags) {
    wmma_gemm_h_kernel<<<dim3(Nd / 256, M / 64), 256, 0, stream>>>(
        A, WT, B, R, Co, Co16, M, Kd, Nd, flags);
  };

  // ---- embedding ----
  embed_kernel<<<S_LEN, 256, 0, stream>>>(ids, tok, pos, X);

  // ---- transformer layers ----
  for (int L = 0; L < 6; ++L) {
    const int b = 3 + L * 16;
    const float *qbias = F(b + 1), *kbias = F(b + 3), *vbias = F(b + 5);
    const float *obias = F(b + 7), *f1b = F(b + 9), *f2b = F(b + 11);
    const float *g1n = F(b + 12), *b1n = F(b + 13);
    const float *g2n = F(b + 14), *b2n = F(b + 15);

    layernorm_kernel<<<S_LEN, 32, 0, stream>>>(X, g1n, b1n, X2h);
    gemm(X2h, qwT[L], qbias, nullptr, nullptr, Qh,  S_LEN, HIDC, HIDC, 4);
    gemm(X2h, kwT[L], kbias, nullptr, nullptr, Kh,  S_LEN, HIDC, HIDC, 4);
    gemm(X2h, vwT[L], vbias, nullptr, nullptr, Vth, S_LEN, HIDC, HIDC, 8); // transposed
    attn_band_kernel<<<dim3(S_LEN / 16, NHEAD), 32, 0, stream>>>(Qh, Kh, Vth, Ch);
    gemm(Ch, owT[L], obias, X, X, nullptr, S_LEN, HIDC, HIDC, 2);        // +resid
    layernorm_kernel<<<S_LEN, 32, 0, stream>>>(X, g2n, b2n, X2h);
    gemm(X2h, f1T[L], f1b, nullptr, nullptr, FFh, S_LEN, HIDC, 4 * HIDC, 1 | 4);
    gemm(FFh, f2T[L], f2b, X, X, Xh, S_LEN, 4 * HIDC, HIDC, 2 | 4);      // +resid
  }

  // ---- memory module ----
  gemm(Xh, rqT, rqb, nullptr, nullptr, Qh, S_LEN, HIDC, MDIM_N, 4);      // rq (f16)
  gemm(Qh, memNT, nullptr, nullptr, MS, nullptr, S_LEN, MDIM_N, SLOTS_N, 2);
  row_softmax_kernel<<<S_LEN, 32, 0, stream>>>(MS, MSh, SLOTS_N);
  gemm(MSh, memT, nullptr, nullptr, RD, nullptr, S_LEN, SLOTS_N, MDIM_N, 2);
  concat_kernel<<<S_LEN, 256, 0, stream>>>(X, RD, CATh);
  gemm(CATh, moutT, moutb, nullptr, nullptr, Sh0, S_LEN, 2 * HIDC, HIDC, 4);

  // ---- ACT loop (10 fixed steps) ----
  hipMemsetAsync(CbF, 0, (size_t)S_LEN * HIDC * sizeof(float), stream);
  hipMemsetAsync(HP, 0, (size_t)S_LEN * sizeof(float), stream);
  hipMemsetAsync(NU, 0, (size_t)S_LEN * sizeof(float), stream);
  const _Float16* state = Sh0;
  _Float16* nextA = Sh1; _Float16* nextB = Sh0;
  for (int step = 0; step < 10; ++step) {
    gemm(state, h1T, h1b, nullptr, G1, nullptr, S_LEN, HIDC, 256, 1 | 2);
    act_halt_kernel<<<S_LEN, 32, 0, stream>>>(G1, h2w, h2b, HP, NU, PA);
    gemm(state, p1T, p1b, nullptr, nullptr, FFh, S_LEN, HIDC, 2 * HIDC, 1 | 4);
    gemm(FFh, p2T, p2b, nullptr, PROC, nextA, S_LEN, 2 * HIDC, HIDC, 2 | 4);
    act_accum_kernel<<<S_LEN, 256, 0, stream>>>(PROC, PA, CbF);
    state = nextA; _Float16* t = nextA; nextA = nextB; nextB = t;
  }

  // ---- logits + ponder ----
  cvt_f16_kernel<<<(S_LEN * HIDC + 255) / 256, 256, 0, stream>>>(
      CbF, Cbh, S_LEN * HIDC);
  float* logits = (float*)d_out;
  gemm(Cbh, outT, outb, nullptr, logits, nullptr, S_LEN, HIDC, VOCAB_N, 2);
  ponder_kernel<<<1, 256, 0, stream>>>(NU, logits + (size_t)S_LEN * VOCAB_N);
}
